// GatedGAT_4861902979304
// MI455X (gfx1250) — compile-verified, matrix operand order checked by reading, not run on hardware
//
#include <hip/hip_runtime.h>

#define BB 16
#define NN 1024
#define FF 128
#define NEGV -9.0e15f

typedef __attribute__((ext_vector_type(16))) __bf16 v16bf;
typedef __attribute__((ext_vector_type(8)))  __bf16 v8bf;
typedef __attribute__((ext_vector_type(8)))  float  v8f;

// Load a WMMA 16-bit operand (A or B side of X @ Yt, or B side from a
// transposed matrix) from a row-major bf16 row pointer.
// ISA 7.12.2 16-bit 16x32 layout: lanes 0-15 get K = c*32 + {0..7, 16..23},
// lanes 16-31 get K = c*32 + {8..15, 24..31}.
__device__ inline v16bf load_row_op(const __bf16* p, int c, int half) {
  const __bf16* q = p + c * 32 + (half ? 8 : 0);
  v8bf lo = *(const v8bf*)(q);        // K+0 .. K+7   (16 bytes)
  v8bf hi = *(const v8bf*)(q + 16);   // K+16 .. K+23 (16 bytes)
  v16bf v;
#pragma unroll
  for (int i = 0; i < 8; ++i) { v[i] = lo[i]; v[8 + i] = hi[i]; }
  return v;
}

__device__ inline float redmax16(float v) {
#pragma unroll
  for (int m = 1; m <= 8; m <<= 1) v = fmaxf(v, __shfl_xor(v, m, 32));
  return v;
}
__device__ inline float redsum16(float v) {
#pragma unroll
  for (int m = 1; m <= 8; m <<= 1) v += __shfl_xor(v, m, 32);
  return v;
}

// Pack W (128x128) and A (128x128) f32 row-major into bf16 B-operand order:
// blob[((n_chunk*4 + k_chunk)*32 + lane)*16 + i]
__global__ void pack_mats(const float* __restrict__ W, const float* __restrict__ A,
                          __bf16* __restrict__ Wpk, __bf16* __restrict__ Apk) {
  int t = blockIdx.x * blockDim.x + threadIdx.x;   // 0..16383
  if (t >= FF * FF) return;
  int i    = t & 15;
  int lane = (t >> 4) & 31;
  int kc   = (t >> 9) & 3;
  int nc   = (t >> 11) & 7;
  int half = lane >> 4;
  int col  = nc * 16 + (lane & 15);
  int K    = i + (half ? 8 : 0) + (i >= 8 ? 8 : 0);
  int row  = kc * 32 + K;
  Wpk[t] = (__bf16)W[row * FF + col];
  Apk[t] = (__bf16)A[row * FF + col];
}

// Per 16-row tile: h = x@W + b ; g = h@A. Emit h (row-major bf16),
// h transposed per batch (for PV B-operand), g (row-major bf16).
__global__ __launch_bounds__(32) void k_hg(
    const float* __restrict__ x, const float* __restrict__ Wb,
    const __bf16* __restrict__ Wpk, const __bf16* __restrict__ Apk,
    __bf16* __restrict__ h_bf16, __bf16* __restrict__ hT,
    __bf16* __restrict__ g_bf16) {
  const int lane = threadIdx.x & 31;
  const int half = lane >> 4;
  const int lr   = lane & 15;
  const int row0 = blockIdx.x * 16;        // global row in [0, B*N)
  const int b    = row0 >> 10;
  const int n0   = row0 & (NN - 1);

  __shared__ __bf16 lds_h[16 * FF];

  // x tile -> A operands (f32 -> bf16), lane lr holds row lr of the tile
  v16bf ax[4];
  const float* xrow = x + (size_t)(row0 + lr) * FF;
#pragma unroll
  for (int c = 0; c < 4; ++c) {
    const float* q = xrow + c * 32 + (half ? 8 : 0);
#pragma unroll
    for (int i = 0; i < 8; ++i) { ax[c][i] = (__bf16)q[i]; ax[c][8 + i] = (__bf16)q[16 + i]; }
  }

  // h = x @ W + bias    (8 n-chunks x 4 K-chunks)
  v8f hc[8];
#pragma unroll
  for (int n = 0; n < 8; ++n) {
    v8f c = {};
#pragma unroll
    for (int k = 0; k < 4; ++k) {
      v16bf bw = *(const v16bf*)(Wpk + ((size_t)((n * 4 + k) * 32 + lane) * 16));
      c = __builtin_amdgcn_wmma_f32_16x16x32_bf16(false, ax[k], false, bw,
                                                  (short)0, c, false, false);
    }
    float bias = Wb[n * 16 + lr];
#pragma unroll
    for (int r = 0; r < 8; ++r) c[r] += bias;
    hc[n] = c;
  }

  // store h: row-major bf16 + LDS (scalar), transposed via packed b128
#pragma unroll
  for (int n = 0; n < 8; ++n) {
    const int col = n * 16 + lr;
    v8bf tv;
#pragma unroll
    for (int r = 0; r < 8; ++r) {
      const int row = r + half * 8;
      __bf16 vb = (__bf16)hc[n][r];
      tv[r] = vb;
      h_bf16[(size_t)(row0 + row) * FF + col] = vb;
      lds_h[row * FF + col] = vb;
    }
    // lane holds 8 consecutive rows (half*8 .. half*8+7) of column `col`
    *(v8bf*)(hT + ((size_t)b * FF + col) * NN + n0 + half * 8) = tv;
  }
  __syncthreads();

  // reload h tile in A-operand layout
  v16bf ah[4];
#pragma unroll
  for (int c = 0; c < 4; ++c) ah[c] = load_row_op(lds_h + lr * FF, c, half);

  // g = h @ A
#pragma unroll
  for (int n = 0; n < 8; ++n) {
    v8f c = {};
#pragma unroll
    for (int k = 0; k < 4; ++k) {
      v16bf ba = *(const v16bf*)(Apk + ((size_t)((n * 4 + k) * 32 + lane) * 16));
      c = __builtin_amdgcn_wmma_f32_16x16x32_bf16(false, ah[k], false, ba,
                                                  (short)0, c, false, false);
    }
    const int col = n * 16 + lr;
#pragma unroll
    for (int r = 0; r < 8; ++r)
      g_bf16[(size_t)(row0 + r + half * 8) * FF + col] = (__bf16)c[r];
  }
}

// Fused: e = G_j.Hk^T + H_j.Gk^T, mask by adj, online softmax, PV into
// resident f32 accumulator, then relu + sigmoid gate + blend -> out.
__global__ __launch_bounds__(32) void k_attn(
    const __bf16* __restrict__ hmat, const __bf16* __restrict__ gmat,
    const __bf16* __restrict__ hT, const float* __restrict__ adj,
    const float* __restrict__ x, const float* __restrict__ gate_w,
    const float* __restrict__ gate_b, float* __restrict__ out) {
  const int lane = threadIdx.x & 31;
  const int half = lane >> 4;
  const int lr   = lane & 15;
  const int b    = blockIdx.x >> 6;
  const int j0   = (blockIdx.x & 63) * 16;

  __shared__ __bf16 lds_p[16 * 32];   // C-layout -> A-layout bounce for P

  // resident A-operands for this j-tile
  v16bf aG[4], aH[4];
  const __bf16* grow = gmat + (size_t)(b * NN + j0 + lr) * FF;
  const __bf16* hrow = hmat + (size_t)(b * NN + j0 + lr) * FF;
#pragma unroll
  for (int c = 0; c < 4; ++c) { aG[c] = load_row_op(grow, c, half); aH[c] = load_row_op(hrow, c, half); }

  // all-ones B operand: row-sum of P via WMMA (l = P @ 1)
  v16bf ones;
#pragma unroll
  for (int i = 0; i < 16; ++i) ones[i] = (__bf16)1.0f;

  // online-softmax state per C-layout row slot (slot r -> row r + half*8)
  float m[8];
#pragma unroll
  for (int r = 0; r < 8; ++r) m[r] = -3.0e38f;
  v8f zero = {};
  v8f acc[8], acc_l = zero;
#pragma unroll
  for (int n = 0; n < 8; ++n) acc[n] = zero;

  for (int k0 = 0; k0 < NN; k0 += 32) {
    // ---- logits: two 16x16 tiles over K=128 features
    v8f e[2];
#pragma unroll
    for (int sub = 0; sub < 2; ++sub) {
      const __bf16* hk = hmat + (size_t)(b * NN + k0 + sub * 16 + lr) * FF;
      const __bf16* gk = gmat + (size_t)(b * NN + k0 + sub * 16 + lr) * FF;
      v8f c = {};
#pragma unroll
      for (int cc = 0; cc < 4; ++cc) {
        v16bf bh = load_row_op(hk, cc, half);
        c = __builtin_amdgcn_wmma_f32_16x16x32_bf16(false, aG[cc], false, bh,
                                                    (short)0, c, false, false);
      }
#pragma unroll
      for (int cc = 0; cc < 4; ++cc) {
        v16bf bg = load_row_op(gk, cc, half);
        c = __builtin_amdgcn_wmma_f32_16x16x32_bf16(false, aH[cc], false, bg,
                                                    (short)0, c, false, false);
      }
      e[sub] = c;
    }

    // ---- mask with adj
    float av[2][8], em[2][8];
#pragma unroll
    for (int sub = 0; sub < 2; ++sub) {
      const int col = k0 + sub * 16 + lr;
#pragma unroll
      for (int r = 0; r < 8; ++r)
        av[sub][r] = adj[((size_t)b * NN + j0 + r + half * 8) * NN + col];
    }
    float mt[8];
    bool upd = false;
#pragma unroll
    for (int r = 0; r < 8; ++r) {
      em[0][r] = (av[0][r] > 1e-6f) ? e[0][r] : NEGV;
      em[1][r] = (av[1][r] > 1e-6f) ? e[1][r] : NEGV;
      mt[r] = redmax16(fmaxf(em[0][r], em[1][r]));
      upd = upd || (mt[r] > m[r]);
    }
    // ---- rescale only when some row max actually increased (wave-uniform)
    if (__ballot(upd) != 0ull) {
#pragma unroll
      for (int r = 0; r < 8; ++r) {
        float mn = fmaxf(m[r], mt[r]);
        float scale = __expf(m[r] - mn);
        m[r] = mn;
        acc_l[r] *= scale;
#pragma unroll
        for (int n = 0; n < 8; ++n) acc[n][r] *= scale;
      }
    }
    // ---- probabilities (attention *= adj folded in) -> LDS in A layout
#pragma unroll
    for (int r = 0; r < 8; ++r) {
      float p0 = (av[0][r] > 1e-6f) ? __expf(em[0][r] - m[r]) : 0.0f;
      float p1 = (av[1][r] > 1e-6f) ? __expf(em[1][r] - m[r]) : 0.0f;
      const int row = r + half * 8;
      lds_p[row * 32 + lr]      = (__bf16)p0;
      lds_p[row * 32 + 16 + lr] = (__bf16)p1;
    }
    __syncthreads();

    // ---- PV: acc += P(16x32) @ H[k0..k0+31, :] ; acc_l += P @ 1
    v16bf aP = load_row_op(lds_p + lr * 32, 0, half);
#pragma unroll
    for (int n = 0; n < 8; ++n) {
      const __bf16* vrow = hT + ((size_t)b * FF + n * 16 + lr) * NN + k0;
      v16bf bv = load_row_op(vrow, 0, half);
      acc[n] = __builtin_amdgcn_wmma_f32_16x16x32_bf16(false, aP, false, bv,
                                                       (short)0, acc[n], false, false);
    }
    acc_l = __builtin_amdgcn_wmma_f32_16x16x32_bf16(false, aP, false, ones,
                                                    (short)0, acc_l, false, false);
    __syncthreads();   // WAR protection on lds_p for next iteration
  }

  // ---- epilogue: h' = relu(acc/l); coeff = sigmoid([x,h'] @ gate_w + gb); blend
  float gw1[8], gw2[8];
#pragma unroll
  for (int n = 0; n < 8; ++n) { gw1[n] = gate_w[n * 16 + lr]; gw2[n] = gate_w[FF + n * 16 + lr]; }
  const float gb = gate_b[0];

#pragma unroll
  for (int r = 0; r < 8; ++r) {
    const int row = j0 + r + half * 8;
    const size_t base = ((size_t)b * NN + row) * FF;
    float linv = 1.0f / acc_l[r];   // row sum, replicated across columns
    float hp[8], xr[8];
    float dot = 0.0f;
#pragma unroll
    for (int n = 0; n < 8; ++n) {
      hp[n] = fmaxf(acc[n][r] * linv, 0.0f);
      xr[n] = x[base + n * 16 + lr];
      dot += xr[n] * gw1[n] + hp[n] * gw2[n];
    }
    dot = redsum16(dot);
    float coeff = 1.0f / (1.0f + __expf(-(dot + gb)));
#pragma unroll
    for (int n = 0; n < 8; ++n)
      out[base + n * 16 + lr] = coeff * xr[n] + (1.0f - coeff) * hp[n];
  }
}

extern "C" void kernel_launch(void* const* d_in, const int* in_sizes, int n_in,
                              void* d_out, int out_size, void* d_ws, size_t ws_size,
                              hipStream_t stream) {
  const float* x      = (const float*)d_in[0];
  const float* adj    = (const float*)d_in[1];
  const float* W_w    = (const float*)d_in[2];
  const float* W_b    = (const float*)d_in[3];
  const float* A      = (const float*)d_in[4];
  const float* gate_w = (const float*)d_in[5];
  const float* gate_b = (const float*)d_in[6];
  float* out = (float*)d_out;

  char* ws = (char*)d_ws;
  __bf16* Wpk    = (__bf16*)ws; ws += (size_t)FF * FF * 2;
  __bf16* Apk    = (__bf16*)ws; ws += (size_t)FF * FF * 2;
  __bf16* h_bf16 = (__bf16*)ws; ws += (size_t)BB * NN * FF * 2;
  __bf16* hT     = (__bf16*)ws; ws += (size_t)BB * NN * FF * 2;
  __bf16* g_bf16 = (__bf16*)ws; ws += (size_t)BB * NN * FF * 2;

  pack_mats<<<(FF * FF + 255) / 256, 256, 0, stream>>>(W_w, A, Wpk, Apk);
  k_hg<<<BB * NN / 16, 32, 0, stream>>>(x, W_b, Wpk, Apk, h_bf16, hT, g_bf16);
  k_attn<<<BB * (NN / 16), 32, 0, stream>>>(h_bf16, g_bf16, hT, adj, x,
                                            gate_w, gate_b, out);
}
